// DynamicsCore_12481174962768
// MI455X (gfx1250) — matrix operand order repaired and checked by reading, not compiled, not run on hardware
//
#include <hip/hip_runtime.h>
#include <hip/hip_bf16.h>
#include <math.h>

// ---------------------------------------------------------------------------
// DynamicsCore for MI455X (gfx1250, wave32, WMMA).
// Persistent-kernel implementation: 64 workgroups x 256 threads (8 waves),
// software grid barrier between dependent stages. All 16-row GEMMs run on
// v_wmma_f32_16x16x32_bf16 (bf16 A/B converted on the fly / pre-converted
// weights, f32 accumulate). Element-wise field/RK4 math stays f32.
// ---------------------------------------------------------------------------

typedef __bf16 bf16_t;
typedef __attribute__((ext_vector_type(16))) __bf16 v16bf;
typedef __attribute__((ext_vector_type(8)))  float  v8f;

#define NWG        64
#define BLOCK      256
#define NWAVES     512          // NWG * BLOCK/32
#define NTHREADS   16384        // NWG * BLOCK
#define D_         1024
#define B_         16
#define S_         32
#define WIN_       16
#define DT_        0.25f

// ---- bf16 weight copy: element offsets inside d_ws ------------------------
static constexpr size_t OW_Wd   = 0;
static constexpr size_t OW_Wu   = 262144;
static constexpr size_t OW_Wi   = 524288;
static constexpr size_t OW_Wm   = 1572864;
static constexpr size_t OW_Wc   = 2621440;
static constexpr size_t OW_gin  = 3670016;   // 2048x1024
static constexpr size_t OW_gmem = 5767168;
static constexpr size_t OW_gctx = 7864320;
static constexpr size_t OW_res  = 9961472;
static constexpr size_t OW_Wq   = 12058624;
static constexpr size_t OW_Wk   = 13107200;
static constexpr size_t OW_Wv   = 14155776;
static constexpr size_t OW_Wo   = 15204352;
static constexpr size_t OW_cg   = 16252928;  // 2048x1024
static constexpr size_t NW_TOT  = 18350080;  // total bf16 elements
static constexpr size_t FB_BYTES = NW_TOT * 2;   // 36,700,160 (256B aligned)

// ---- f32 scratch: float offsets after the bf16 block -----------------------
static constexpr size_t OF_x    = 0;        // state x           [16][1024]
static constexpr size_t OF_sev  = 16384;    // RK4 eval point / xtmp
static constexpr size_t OF_acc  = 32768;    // RK4 k-accumulator
static constexpr size_t OF_q    = 49152;    // x@Wq
static constexpr size_t OF_zcgx = 65536;    // x@cg_W[:D]
static constexpr size_t OF_pin  = 81920;    // u@Wi
static constexpr size_t OF_pmem = 98304;    // m@Wm
static constexpr size_t OF_pctx = 114688;   // c@Wc
static constexpr size_t OF_ain  = 131072;   // u@gin_W[D:]
static constexpr size_t OF_amem = 147456;   // m@gmem_W[D:]
static constexpr size_t OF_actx = 163840;   // c@gctx_W[D:]
static constexpr size_t OF_ares = 180224;   // u@res_W[D:]
static constexpr size_t OF_ctx  = 196608;   // attention ctx (pre-Wo)
static constexpr size_t OF_ctx2 = 212992;   // ctx@Wo
static constexpr size_t OF_c    = 229376;   // gated coupling output
static constexpr size_t OF_zin  = 245760;   // s@gin_W[:D]
static constexpr size_t OF_zmem = 262144;
static constexpr size_t OF_zctx = 278528;
static constexpr size_t OF_zres = 294912;
static constexpr size_t OF_zd   = 311296;   // s@Wd   [16][256]
static constexpr size_t OF_kc   = 315392;   // K cache [WIN][16][1024]
static constexpr size_t OF_vc   = 577536;   // V cache
static constexpr size_t OF_ss   = 839680;   // per-row sum-of-squares [16]
static constexpr size_t OF_cnt  = 839696;   // barrier counter (int)
// total ws: FB_BYTES + ~839712*4  ~= 40.1 MB

struct Params {
  const float *x0, *imp, *mem;
  const float *Wd, *Wu, *Wi, *Wm, *Wc;
  const float *gin_W, *gin_b, *gmem_W, *gmem_b, *gctx_W, *gctx_b, *res_W, *res_b, *fbias;
  const float *Wq, *Wk, *Wv, *Wo, *cg_W, *cg_b, *gamma;
  const float *log_damping, *log_tau_fast, *log_tau_slow;
};

__device__ __forceinline__ float sigm(float z) { return 1.f / (1.f + __expf(-z)); }

__device__ __forceinline__ float wred32(float v) {
#pragma unroll
  for (int m = 16; m > 0; m >>= 1) v += __shfl_xor(v, m, 32);
  return v;
}
__device__ __forceinline__ float wred16(float v) {   // reduce within 16-lane halves
#pragma unroll
  for (int m = 8; m > 0; m >>= 1) v += __shfl_xor(v, m, 32);
  return v;
}

// C = A(16xK f32, row stride lda) x W(KxN bf16, row stride ldw), column tile n0.
// A-fragment (16x32 bf16): lane m=L&15, half=L>>4; e -> k = (e&7)+8*half+16*(e>>3)
// B-fragment (32x16 bf16): lane n=L&15, half=L>>4; e -> k = (e&7)+8*half+16*(e>>3)
template <bool TANH>
__device__ __forceinline__ v8f gemm_tile(const float* A, int lda,
                                         const bf16_t* W, int ldw,
                                         int K, int n0) {
  const int lane = threadIdx.x & 31;
  const int mrow = lane & 15, half = lane >> 4;
  v8f cacc = {0.f, 0.f, 0.f, 0.f, 0.f, 0.f, 0.f, 0.f};
  const float* arow = A + (size_t)mrow * lda + 8 * half;
  for (int kb = 0; kb < K; kb += 32) {
    v8f f0 = *(const v8f*)(arow + kb);        // k = kb + 8*half + e      (e<8)
    v8f f1 = *(const v8f*)(arow + kb + 16);   // k = kb + 16 + 8*half + e (e>=8)
    v16bf a;
#pragma unroll
    for (int e = 0; e < 8; e++) { float v = f0[e]; if (TANH) v = tanhf(v); a[e] = (bf16_t)v; }
#pragma unroll
    for (int e = 0; e < 8; e++) { float v = f1[e]; if (TANH) v = tanhf(v); a[8 + e] = (bf16_t)v; }
    v16bf b;
#pragma unroll
    for (int e = 0; e < 16; e++) b[e] = W[(size_t)(kb + (e & 7) + 8 * half + 16 * (e >> 3)) * ldw + n0 + mrow];
    cacc = __builtin_amdgcn_wmma_f32_16x16x32_bf16(false, a, false, b,
                                                   (short)0, cacc, false, false);
  }
  return cacc;
}

// D-tile store: lane n=L&15; element e -> row m = e + 8*(L>>4)
__device__ __forceinline__ void store_tile(float* O, int ldo, int n0, v8f c) {
  const int lane = threadIdx.x & 31;
  const int n = lane & 15, half = lane >> 4;
#pragma unroll
  for (int e = 0; e < 8; e++) O[(size_t)(e + 8 * half) * ldo + n0 + n] = c[e];
}

// ---------------------------------------------------------------------------
// Init: convert weights f32 -> bf16 copy in ws, copy x0, zero acc/KV/ctl.
// ---------------------------------------------------------------------------
__global__ void init_kernel(Params p, char* ws) {
  bf16_t* wbf = (bf16_t*)ws;
  float*  fbu = (float*)(ws + FB_BYTES);
  const size_t stride = (size_t)gridDim.x * blockDim.x;
  const size_t i0 = (size_t)blockIdx.x * blockDim.x + threadIdx.x;
  const float* srcs[14] = {p.Wd, p.Wu, p.Wi, p.Wm, p.Wc, p.gin_W, p.gmem_W,
                           p.gctx_W, p.res_W, p.Wq, p.Wk, p.Wv, p.Wo, p.cg_W};
  const size_t offs[15] = {OW_Wd, OW_Wu, OW_Wi, OW_Wm, OW_Wc, OW_gin, OW_gmem,
                           OW_gctx, OW_res, OW_Wq, OW_Wk, OW_Wv, OW_Wo, OW_cg, NW_TOT};
  for (size_t i = i0; i < NW_TOT; i += stride) {
    int r = 0;
    while (i >= offs[r + 1]) ++r;
    wbf[i] = (bf16_t)srcs[r][i - offs[r]];
  }
  for (size_t i = i0; i < 16384; i += stride) fbu[OF_x + i] = p.x0[i];
  for (size_t i = i0; i < 16384; i += stride) fbu[OF_acc + i] = 0.f;
  for (size_t i = i0; i < 524288; i += stride) fbu[OF_kc + i] = 0.f;  // kc+vc
  for (size_t i = i0; i < 32; i += stride) fbu[OF_ss + i] = 0.f;      // ss+counter
}

// ---------------------------------------------------------------------------
// Persistent dynamics kernel.
// ---------------------------------------------------------------------------
__global__ __launch_bounds__(BLOCK, 1) void dyn_kernel(Params p, char* ws, float* out) {
  bf16_t* wbf = (bf16_t*)ws;
  float*  fbu = (float*)(ws + FB_BYTES);
  float *x = fbu + OF_x, *sev = fbu + OF_sev, *acc = fbu + OF_acc;
  float *q = fbu + OF_q, *zcgx = fbu + OF_zcgx;
  float *pin = fbu + OF_pin, *pmem = fbu + OF_pmem, *pctx = fbu + OF_pctx;
  float *ain = fbu + OF_ain, *amem = fbu + OF_amem, *actx = fbu + OF_actx, *ares = fbu + OF_ares;
  float *fctx = fbu + OF_ctx, *ctx2 = fbu + OF_ctx2, *cbuf = fbu + OF_c;
  float *zin = fbu + OF_zin, *zmem = fbu + OF_zmem, *zctx = fbu + OF_zctx, *zres = fbu + OF_zres;
  float *zd = fbu + OF_zd, *kcache = fbu + OF_kc, *vcache = fbu + OF_vc, *ss = fbu + OF_ss;
  int* cnt = (int*)(fbu + OF_cnt);

  const bf16_t *wd = wbf + OW_Wd, *wu = wbf + OW_Wu, *wi = wbf + OW_Wi;
  const bf16_t *wm = wbf + OW_Wm, *wc = wbf + OW_Wc;
  const bf16_t *wgin = wbf + OW_gin, *wgmem = wbf + OW_gmem, *wgctx = wbf + OW_gctx;
  const bf16_t *wres = wbf + OW_res, *wq = wbf + OW_Wq, *wk = wbf + OW_Wk;
  const bf16_t *wv = wbf + OW_Wv, *wo = wbf + OW_Wo, *wcg = wbf + OW_cg;

  const int tid = threadIdx.x, lane = tid & 31;
  const int gw = blockIdx.x * 8 + (tid >> 5);        // global wave id [0,512)
  const int gtid = blockIdx.x * BLOCK + tid;         // [0,16384)
  int gen = 0;

  const float lam = __expf(p.log_damping[0]);
  const float itf = __expf(-p.log_tau_fast[0]);
  const float its = __expf(-p.log_tau_slow[0]);

#define GSYNC()                                                         \
  do {                                                                  \
    gen++;                                                              \
    __threadfence();                                                    \
    __syncthreads();                                                    \
    if (tid == 0) {                                                     \
      atomicAdd(cnt, 1);                                                \
      while (((volatile int*)cnt)[0] < gen * NWG)                       \
        __builtin_amdgcn_s_sleep(1);                                    \
    }                                                                   \
    __syncthreads();                                                    \
    __threadfence();                                                    \
  } while (0)

  for (int t = 0; t < S_; ++t) {
    const float* u  = p.imp + (size_t)t * D_;   // row stride S*D
    const float* mm = p.mem + (size_t)t * D_;
    const int slot = (t - 1) & (WIN_ - 1);      // K/V slot for y_{t-1}

    // ---- Stage A: x/u/m projections (+ append y_{t-1} into K/V caches) ----
    const int nA = (t > 0 ? 9 : 7) * 64;
    for (int j = gw; j < nA; j += NWAVES) {
      const int mat = j >> 6, n0 = (j & 63) * 16;
      const float* A; int lda; const bf16_t* W; float* O;
      switch (mat) {
        case 0:  A = x;  lda = D_;    W = wq;              O = q;    break;
        case 1:  A = x;  lda = D_;    W = wcg;             O = zcgx; break;
        case 2:  A = u;  lda = 32768; W = wi;              O = pin;  break;
        case 3:  A = mm; lda = 32768; W = wm;              O = pmem; break;
        case 4:  A = u;  lda = 32768; W = wgin + 1048576;  O = ain;  break;
        case 5:  A = mm; lda = 32768; W = wgmem + 1048576; O = amem; break;
        case 6:  A = u;  lda = 32768; W = wres + 1048576;  O = ares; break;
        case 7:  A = x;  lda = D_;    W = wk;              O = kcache + slot * 16384; break;
        default: A = x;  lda = D_;    W = wv;              O = vcache + slot * 16384; break;
      }
      store_tile(O, D_, n0, gemm_tile<false>(A, lda, W, D_, D_, n0));
    }
    GSYNC();

    // ---- Stage B: window attention, one wave per (b,h) --------------------
    for (int j = gw; j < 128; j += NWAVES) {
      const int b = j >> 3, h = j & 7;
      const int d0 = h * 128 + lane * 4;        // each lane owns 4 dims of DH
      const float* qb = q + b * D_ + d0;
      const float q0 = qb[0], q1 = qb[1], q2 = qb[2], q3 = qb[3];
      float sv[WIN_];
      float mx = -3.4e38f;
      for (int w = 0; w < WIN_; w++) {
        const float* kb = kcache + w * 16384 + b * D_ + d0;
        float d = q0 * kb[0] + q1 * kb[1] + q2 * kb[2] + q3 * kb[3];
        d = wred32(d) * 0.08838834764831845f;   // 1/sqrt(128)
        sv[w] = d;
        mx = fmaxf(mx, d);
      }
      float den = 0.f;
      for (int w = 0; w < WIN_; w++) { float e = __expf(sv[w] - mx); sv[w] = e; den += e; }
      const float inv = 1.f / den;
      float c0 = 0.f, c1 = 0.f, c2 = 0.f, c3 = 0.f;
      for (int w = 0; w < WIN_; w++) {
        const float* vb = vcache + w * 16384 + b * D_ + d0;
        c0 += sv[w] * vb[0]; c1 += sv[w] * vb[1]; c2 += sv[w] * vb[2]; c3 += sv[w] * vb[3];
      }
      float* cb = fctx + b * D_ + d0;
      cb[0] = c0 * inv; cb[1] = c1 * inv; cb[2] = c2 * inv; cb[3] = c3 * inv;
    }
    GSYNC();

    // ---- Stage C: ctx2 = ctx @ Wo -----------------------------------------
    for (int j = gw; j < 64; j += NWAVES)
      store_tile(ctx2, D_, j * 16, gemm_tile<false>(fctx, D_, wo, D_, D_, j * 16));
    GSYNC();

    // ---- Stage D: z2 = ctx2 @ cg_W[D:]; c = sigmoid(zcgx+z2+cg_b)*ctx2 ----
    for (int j = gw; j < 64; j += NWAVES) {
      const int n0 = j * 16;
      v8f z2 = gemm_tile<false>(ctx2, D_, wcg + 1048576, D_, D_, n0);
      const int n = lane & 15, half = lane >> 4, col = n0 + n;
      const float cgb = p.cg_b[col];
#pragma unroll
      for (int e = 0; e < 8; e++) {
        const int o = (e + 8 * half) * D_ + col;
        cbuf[o] = sigm(zcgx[o] + z2[e] + cgb) * ctx2[o];
      }
    }
    GSYNC();

    // ---- Stage F: c-dependent per-step terms ------------------------------
    for (int j = gw; j < 128; j += NWAVES) {
      const int n0 = (j & 63) * 16;
      if (j < 64) store_tile(pctx, D_, n0, gemm_tile<false>(cbuf, D_, wc, D_, D_, n0));
      else        store_tile(actx, D_, n0, gemm_tile<false>(cbuf, D_, wgctx + 1048576, D_, D_, n0));
    }
    GSYNC();

    // ---- RK4 substeps ------------------------------------------------------
    for (int sub = 0; sub < 4; ++sub) {
      for (int i = 0; i < 4; ++i) {
        const float* s_src = (i == 0) ? x : sev;
        if (i == 3 && gtid < 16) ss[gtid] = 0.f;   // reset sumsq before eval-4 atomics

        // eval stage 1: state-dependent matmuls (gates + Zd)
        for (int j = gw; j < 272; j += NWAVES) {
          if (j < 256) {
            const int mat = j >> 6, n0 = (j & 63) * 16;
            const bf16_t* W; float* O;
            switch (mat) {
              case 0:  W = wgin;  O = zin;  break;
              case 1:  W = wgmem; O = zmem; break;
              case 2:  W = wgctx; O = zctx; break;
              default: W = wres;  O = zres; break;
            }
            store_tile(O, D_, n0, gemm_tile<false>(s_src, D_, W, D_, D_, n0));
          } else {
            const int n0 = (j - 256) * 16;
            store_tile(zd, 256, n0, gemm_tile<false>(s_src, D_, wd, 256, D_, n0));
          }
        }
        GSYNC();

        // eval stage 2: sc = tanh(Zd)@Wu fused with element-wise field + RK4
        for (int j = gw; j < 64; j += NWAVES) {
          const int n0 = j * 16;
          v8f scc = gemm_tile<true>(zd, 256, wu, D_, 256, n0);
          const int n = lane & 15, half = lane >> 4, col = n0 + n;
          const float ginb = p.gin_b[col], gmb = p.gmem_b[col];
          const float gcb = p.gctx_b[col], rsb = p.res_b[col], fbv = p.fbias[col];
          const float itau = (col < 512) ? itf : its;
          const float wacc = (i == 1 || i == 2) ? 2.f : 1.f;
          const float cstep = (i < 2 ? 0.5f : 1.f) * DT_;
#pragma unroll
          for (int e = 0; e < 8; e++) {
            const int mr = e + 8 * half;
            const int o = mr * D_ + col;
            const float gi = sigm(zin[o] + ain[o] + ginb);
            const float gm = sigm(zmem[o] + amem[o] + gmb);
            const float gc = sigm(zctx[o] + actx[o] + gcb);
            const float rr = sigm(zres[o] + ares[o] + rsb);
            const float pre = scc[e] + gi * pin[o] + gm * pmem[o] + gc * pctx[o] + fbv;
            const float svl = s_src[o];
            const float fv = rr * svl + (1.f - rr) * tanhf(pre);
            const float kv = (-lam * svl + fv) * itau;
            const float a = acc[o] + wacc * kv;
            acc[o] = a;
            if (i < 3) {
              sev[o] = x[o] + cstep * kv;
            } else {
              const float xt = x[o] + (DT_ / 6.f) * a;
              sev[o] = xt;                       // xtmp for finalize
              const float v = wred16(xt * xt);   // reduce over 16 columns
              if ((lane & 15) == 0) atomicAdd(&ss[mr], v);
            }
          }
        }
        GSYNC();
      }

      // finalize substep: norm clamp (+ RMS state-norm and output on sub==3)
      {
        const int o = gtid;                       // exactly 16384 threads
        const int mr = o >> 10, col = o & 1023;
        const float s2 = ss[mr];
        const float scl = fminf(1.f, 10.f / (sqrtf(s2) + 1e-6f));
        float xv = sev[o] * scl;
        if (sub == 3) {
          const float rms = sqrtf(s2 * scl * scl * (1.f / 1024.f) + 1e-6f);
          xv = xv / rms * p.gamma[col];
          out[(size_t)mr * (S_ * D_) + t * D_ + col] = xv;
        }
        x[o] = xv;
        acc[o] = 0.f;
      }
      GSYNC();
    }
  }
#undef GSYNC
}

// ---------------------------------------------------------------------------
extern "C" void kernel_launch(void* const* d_in, const int* in_sizes, int n_in,
                              void* d_out, int out_size, void* d_ws, size_t ws_size,
                              hipStream_t stream) {
  (void)in_sizes; (void)n_in; (void)out_size; (void)ws_size;
  Params p;
  p.x0 = (const float*)d_in[0];
  p.imp = (const float*)d_in[1];
  p.mem = (const float*)d_in[2];
  p.Wd = (const float*)d_in[3];   p.Wu = (const float*)d_in[4];
  p.Wi = (const float*)d_in[5];   p.Wm = (const float*)d_in[6];  p.Wc = (const float*)d_in[7];
  p.gin_W = (const float*)d_in[8];  p.gin_b = (const float*)d_in[9];
  p.gmem_W = (const float*)d_in[10]; p.gmem_b = (const float*)d_in[11];
  p.gctx_W = (const float*)d_in[12]; p.gctx_b = (const float*)d_in[13];
  p.res_W = (const float*)d_in[14];  p.res_b = (const float*)d_in[15];
  p.fbias = (const float*)d_in[16];
  p.Wq = (const float*)d_in[17]; p.Wk = (const float*)d_in[18];
  p.Wv = (const float*)d_in[19]; p.Wo = (const float*)d_in[20];
  p.cg_W = (const float*)d_in[21]; p.cg_b = (const float*)d_in[22];
  p.gamma = (const float*)d_in[23];
  p.log_damping = (const float*)d_in[24];
  p.log_tau_fast = (const float*)d_in[25];
  p.log_tau_slow = (const float*)d_in[26];

  init_kernel<<<512, 256, 0, stream>>>(p, (char*)d_ws);
  dyn_kernel<<<NWG, BLOCK, 0, stream>>>(p, (char*)d_ws, (float*)d_out);
}
